// LoRA_QKV_25013889532113
// MI455X (gfx1250) — compile-verified, matrix operand order checked by reading, not run on hardware
//
#include <hip/hip_runtime.h>

#define DIM   1280
#define NOUT  3840
#define BM    128
#define BN    256
#define BK    64
#define BKP   72            // padded LDS row stride (ushorts): 144B, 16B-aligned
#define NKB   (DIM / BK)    // 20 k-steps

typedef __attribute__((ext_vector_type(16))) __bf16 v16bf;
typedef __attribute__((ext_vector_type(8)))  float  v8f;
typedef __attribute__((ext_vector_type(4)))  int    v4i;

// ---------- CDNA5 async global->LDS (guarded; falls back to reg copies) ----
#if defined(__has_builtin)
#  if __has_builtin(__builtin_amdgcn_global_load_async_to_lds_b128) && \
      __has_builtin(__builtin_amdgcn_s_wait_asynccnt)
#    define USE_ASYNC_LDS 1
#  endif
#endif
#ifndef USE_ASYNC_LDS
#  define USE_ASYNC_LDS 0
#endif

typedef __attribute__((address_space(1))) v4i v4i_g;   // global
typedef __attribute__((address_space(3))) v4i v4i_l;   // LDS

__device__ __forceinline__ void async_b128(const void* g, void* l) {
#if USE_ASYNC_LDS
    __builtin_amdgcn_global_load_async_to_lds_b128((v4i_g*)g, (v4i_l*)l, 0, 0);
#endif
}
__device__ __forceinline__ void async_wait0() {
#if USE_ASYNC_LDS
    __builtin_amdgcn_s_wait_asynccnt(0);
#endif
}

// ---------- fp32 -> bf16 via hardware convert ------------------------------
__device__ __forceinline__ unsigned int cvt2(float lo, float hi) {
    union { __bf16 h[2]; unsigned int u; } p;
    p.h[0] = (__bf16)lo;
    p.h[1] = (__bf16)hi;
    return p.u;
}
__device__ __forceinline__ uint4 pack8(float4 a, float4 b) {
    uint4 o;
    o.x = cvt2(a.x, a.y);  o.y = cvt2(a.z, a.w);
    o.z = cvt2(b.x, b.y);  o.w = cvt2(b.z, b.w);
    return o;
}
__device__ __forceinline__ unsigned short cvt1(float f) {
    union { __bf16 h; unsigned short u; } p;
    p.h = (__bf16)f;
    return p.u;
}

// ---------------------------------------------------------------------------
// Kernel 1: fold LoRA rank-16 update into frozen weight -> bf16 W_eff.
// ---------------------------------------------------------------------------
__global__ __launch_bounds__(256)
void fold_weff(const float* __restrict__ W,
               const float* __restrict__ Aq, const float* __restrict__ Bq,
               const float* __restrict__ Ak, const float* __restrict__ Bk,
               const float* __restrict__ Av, const float* __restrict__ Bv,
               unsigned short* __restrict__ Weff) {
    const int o  = blockIdx.x;
    const int br = o / DIM;
    const int d  = o - br * DIM;
    const float* __restrict__ A = (br == 0) ? Aq : (br == 1) ? Ak : Av;
    const float* __restrict__ B = (br == 0) ? Bq : (br == 1) ? Bk : Bv;

    __shared__ float bs[16];
    if (threadIdx.x < 16) bs[threadIdx.x] = B[d * 16 + threadIdx.x];
    __syncthreads();

    for (int k = threadIdx.x; k < DIM; k += blockDim.x) {
        float acc = W[(size_t)o * DIM + k];
#pragma unroll
        for (int r = 0; r < 16; ++r) acc += bs[r] * A[r * DIM + k];
        Weff[(size_t)o * DIM + k] = cvt1(acc);
    }
}

// ---------------------------------------------------------------------------
// Kernel 2: one-shot x (f32) -> bf16, 8 elements per thread.
// ---------------------------------------------------------------------------
__global__ __launch_bounds__(256)
void cvt_x(const float* __restrict__ x, unsigned short* __restrict__ xbf,
           long long n8) {
    long long i = (long long)blockIdx.x * blockDim.x + threadIdx.x;
    if (i < n8) {
        const float4* p = (const float4*)x + 2 * i;
        ((uint4*)xbf)[i] = pack8(p[0], p[1]);
    }
}

// ---------------------------------------------------------------------------
// Kernel 3: out[M,3840] = x * W_eff^T + bias  (bf16 WMMA, f32 acc)
// 256 threads = 8 waves (2x4), block tile 128x256, wave tile 64x64,
// BK=64, double-buffered LDS, async global->LDS staging.
// ---------------------------------------------------------------------------
template <bool XBF>
__global__ __launch_bounds__(256)
void lora_qkv_gemm(const float* __restrict__ xf,
                   const unsigned short* __restrict__ xbf,
                   const unsigned short* __restrict__ Weff,
                   const float* __restrict__ bias,
                   float* __restrict__ out) {
    __shared__ unsigned short Xs[2][BM * BKP];   // 36 KB
    __shared__ unsigned short Ws[2][BN * BKP];   // 72 KB

    const int tid    = threadIdx.x;
    const int lane   = tid & 31;
    const int wave   = tid >> 5;
    const int wm     = wave & 1;        // 2 wave rows (64 each)
    const int wn     = wave >> 1;       // 4 wave cols (64 each)
    const int laneHi = lane >> 4;
    const int lr     = lane & 15;

    const int mblk = blockIdx.y * BM;
    const int nblk = blockIdx.x * BN;

    // X staging: thread t owns row = t>>1, 32-elem k-segment = (t&1)*32.
    const int xrow = tid >> 1;
    const int xcol = (tid & 1) * 32;
    // W staging: thread t owns row = t, all 64 k.
    const unsigned short* __restrict__ wgT = Weff + (size_t)(nblk + tid) * DIM;
    const unsigned short* __restrict__ xgT = XBF
        ? xbf + (size_t)(mblk + xrow) * DIM + xcol
        : (const unsigned short*)nullptr;
    const float* __restrict__ xfT = XBF
        ? (const float*)nullptr
        : xf + (size_t)(mblk + xrow) * DIM + xcol;

    unsigned short* xdst = &Xs[0][xrow * BKP + xcol];
    unsigned short* wdst = &Ws[0][tid * BKP];
    const int xBufStride = BM * BKP;
    const int wBufStride = BN * BKP;

    auto stage = [&](int kb, int buf) {
        unsigned short* wl = wdst + buf * wBufStride;
        const unsigned short* wgp = wgT + kb * BK;
#if USE_ASYNC_LDS
#pragma unroll
        for (int s = 0; s < 8; ++s) async_b128(wgp + s * 8, wl + s * 8);
#else
#pragma unroll
        for (int s = 0; s < 8; ++s)
            *(uint4*)(wl + s * 8) = *(const uint4*)(wgp + s * 8);
#endif
        unsigned short* xl = xdst + buf * xBufStride;
        if constexpr (XBF) {
            const unsigned short* xgp = xgT + kb * BK;
#if USE_ASYNC_LDS
#pragma unroll
            for (int s = 0; s < 4; ++s) async_b128(xgp + s * 8, xl + s * 8);
#else
#pragma unroll
            for (int s = 0; s < 4; ++s)
                *(uint4*)(xl + s * 8) = *(const uint4*)(xgp + s * 8);
#endif
        } else {
            const float* xgp = xfT + kb * BK;
#pragma unroll
            for (int s = 0; s < 4; ++s) {
                float4 a = ((const float4*)xgp)[2 * s];
                float4 b = ((const float4*)xgp)[2 * s + 1];
                *(uint4*)(xl + s * 8) = pack8(a, b);
            }
        }
    };

    union FragU { uint4 u[2]; v16bf v; };
    // 16-bit A-frag striping: lane<16 -> K 0..7 & 16..23 of row M=lr;
    // lane>=16 -> K 8..15 & 24..31.  W tile stored [n][k] => same pattern for B.
    auto loadA = [&](int buf, int mt, int kh) -> v16bf {
        FragU f;
        const unsigned short* p =
            &Xs[buf][(wm * 64 + mt * 16 + lr) * BKP + kh * 32 + laneHi * 8];
        f.u[0] = *(const uint4*)p;
        f.u[1] = *(const uint4*)(p + 16);
        return f.v;
    };
    auto loadB = [&](int buf, int nt, int kh) -> v16bf {
        FragU f;
        const unsigned short* p =
            &Ws[buf][(wn * 64 + nt * 16 + lr) * BKP + kh * 32 + laneHi * 8];
        f.u[0] = *(const uint4*)p;
        f.u[1] = *(const uint4*)(p + 16);
        return f.v;
    };

    v8f acc[4][4];
#pragma unroll
    for (int mt = 0; mt < 4; ++mt)
#pragma unroll
        for (int nt = 0; nt < 4; ++nt) acc[mt][nt] = {};

    stage(0, 0);
    async_wait0();
    __syncthreads();

    for (int kb = 0; kb < NKB; ++kb) {
        const int cur = kb & 1;
        const int nxt = cur ^ 1;

        if (kb + 1 < NKB) stage(kb + 1, nxt);   // async: fire & forget

#pragma unroll
        for (int kh = 0; kh < 2; ++kh) {
            v16bf afr[4];
#pragma unroll
            for (int mt = 0; mt < 4; ++mt) afr[mt] = loadA(cur, mt, kh);
            // B-frags two at a time to cap live registers (acc 128 + 48).
#pragma unroll
            for (int nh = 0; nh < 2; ++nh) {
                v16bf bfr[2];
#pragma unroll
                for (int b = 0; b < 2; ++b) bfr[b] = loadB(cur, nh * 2 + b, kh);
#pragma unroll
                for (int mt = 0; mt < 4; ++mt)
#pragma unroll
                    for (int b = 0; b < 2; ++b)
                        acc[mt][nh * 2 + b] = __builtin_amdgcn_wmma_f32_16x16x32_bf16(
                            false, afr[mt], false, bfr[b],
                            (short)0, acc[mt][nh * 2 + b], false, false);
            }
        }

        async_wait0();          // own async loads into `nxt` complete
        __syncthreads();        // everyone's complete
    }

    // Epilogue: C/D layout — N = lr (lanes 0-15 / 16-31), M = laneHi*8 + v.
#pragma unroll
    for (int mt = 0; mt < 4; ++mt) {
        const int m0 = mblk + wm * 64 + mt * 16 + laneHi * 8;
#pragma unroll
        for (int nt = 0; nt < 4; ++nt) {
            const int n  = nblk + wn * 64 + nt * 16 + lr;
            const float bv = bias[n];
            float* po = out + (size_t)m0 * NOUT + n;
#pragma unroll
            for (int v = 0; v < 8; ++v)
                po[(size_t)v * NOUT] = acc[mt][nt][v] + bv;
        }
    }
}

// ---------------------------------------------------------------------------
extern "C" void kernel_launch(void* const* d_in, const int* in_sizes, int n_in,
                              void* d_out, int out_size, void* d_ws, size_t ws_size,
                              hipStream_t stream) {
    const float* x   = (const float*)d_in[0];
    const float* Wq  = (const float*)d_in[1];
    const float* bq  = (const float*)d_in[2];
    const float* A_q = (const float*)d_in[3];
    const float* B_q = (const float*)d_in[4];
    const float* A_k = (const float*)d_in[5];
    const float* B_k = (const float*)d_in[6];
    const float* A_v = (const float*)d_in[7];
    const float* B_v = (const float*)d_in[8];
    float* out = (float*)d_out;

    const int M = in_sizes[0] / DIM;                    // 32768

    const size_t needW = (size_t)NOUT * DIM * 2;        // 9.8 MB
    const size_t needX = (size_t)M * DIM * 2;           // 83.9 MB

    unsigned short* Weff = (unsigned short*)d_ws;
    unsigned short* xbf  = (unsigned short*)((char*)d_ws + needW);
    const bool preconv = (ws_size >= needW + needX);

    fold_weff<<<NOUT, 256, 0, stream>>>(Wq, A_q, B_q, A_k, B_k, A_v, B_v, Weff);

    dim3 grid(NOUT / BN, M / BM);
    if (preconv) {
        const long long n8 = (long long)M * DIM / 8;
        cvt_x<<<(unsigned)((n8 + 255) / 256), 256, 0, stream>>>(x, xbf, n8);
        lora_qkv_gemm<true><<<grid, 256, 0, stream>>>(nullptr, xbf, Weff, bq, out);
    } else {
        lora_qkv_gemm<false><<<grid, 256, 0, stream>>>(x, nullptr, Weff, bq, out);
    }
}
